// Net11_88184268521670
// MI455X (gfx1250) — matrix operand order repaired and checked by reading, not compiled, not run on hardware
//
#include <hip/hip_runtime.h>
#include <math.h>

typedef __attribute__((ext_vector_type(16))) _Float16 v16h;
typedef __attribute__((ext_vector_type(8)))  _Float16 v8h;
typedef __attribute__((ext_vector_type(8)))  float    v8f;

__device__ __forceinline__ float silu_f(float v) { return v / (1.0f + __expf(-v)); }

// ---------------------------------------------------------------------------
// Weight convert + transpose: W f32 [K,N] row-major -> Wt f16 [N,K] row-major
// (batched over nmat matrices of identical shape)
// ---------------------------------------------------------------------------
__global__ void k_castT(const float* __restrict__ W, _Float16* __restrict__ Wt,
                        int K, int N, int nmat) {
    long t = (long)blockIdx.x * blockDim.x + threadIdx.x;
    long tot = (long)nmat * K * N;
    if (t >= tot) return;
    int m = (int)(t / ((long)K * N));
    int r = (int)(t - (long)m * K * N);
    int k = r / N, n = r - k * N;
    Wt[(long)m * K * N + (long)n * K + k] = (_Float16)W[t];
}

// ---------------------------------------------------------------------------
// Edge stats: per-edge atomic accumulation of edge_attr into per-node sums
// ---------------------------------------------------------------------------
__global__ void k_estats(const float* __restrict__ ea, const int* __restrict__ col,
                         float* __restrict__ esum, float* __restrict__ esq,
                         float* __restrict__ deg, int E_) {
    int e = blockIdx.x * blockDim.x + threadIdx.x;
    if (e >= E_) return;
    int c = col[e];
    #pragma unroll
    for (int f = 0; f < 6; ++f) {
        float v = ea[(long)e * 6 + f];
        atomicAdd(&esum[(long)c * 6 + f], v);
        atomicAdd(&esq[(long)c * 6 + f], v * v);
    }
    atomicAdd(&deg[c], 1.0f);
}

__global__ void k_x0final(const float* __restrict__ x, const float* __restrict__ esum,
                          const float* __restrict__ esq, const float* __restrict__ deg,
                          float* __restrict__ x0, int N_) {
    int n = blockIdx.x * blockDim.x + threadIdx.x;
    if (n >= N_) return;
    float c = deg[n];
    float* o = x0 + (long)n * 17;
    #pragma unroll
    for (int f = 0; f < 5; ++f) o[f] = x[(long)n * 5 + f];
    #pragma unroll
    for (int f = 0; f < 6; ++f) {
        float s = esum[(long)n * 6 + f];
        float m = s / fmaxf(c, 1.0f);
        float var = (esq[(long)n * 6 + f] - m * s) / fmaxf(c - 1.0f, 1.0f);
        o[5 + f] = m;
        o[11 + f] = sqrtf(fmaxf(var, 0.0f));
    }
}

__global__ void k_ustats(const float* __restrict__ x0, const int* __restrict__ batch,
                         float* __restrict__ usum, float* __restrict__ usq,
                         float* __restrict__ bcnt, int N_) {
    int n = blockIdx.x * blockDim.x + threadIdx.x;
    if (n >= N_) return;
    int b = batch[n];
    #pragma unroll
    for (int f = 0; f < 17; ++f) {
        float v = x0[(long)n * 17 + f];
        atomicAdd(&usum[(long)b * 17 + f], v);
        atomicAdd(&usq[(long)b * 17 + f], v * v);
    }
    atomicAdd(&bcnt[b], 1.0f);
}

__global__ void k_u0final(const float* __restrict__ usum, const float* __restrict__ usq,
                          const float* __restrict__ bcnt, float* __restrict__ u0, int B_) {
    int b = blockIdx.x * blockDim.x + threadIdx.x;
    if (b >= B_) return;
    float c = bcnt[b];
    #pragma unroll
    for (int f = 0; f < 17; ++f) {
        float s = usum[(long)b * 17 + f];
        float m = s / fmaxf(c, 1.0f);
        float var = (usq[(long)b * 17 + f] - m * s) / fmaxf(c - 1.0f, 1.0f);
        u0[(long)b * 34 + f] = m;
        u0[(long)b * 34 + 17 + f] = sqrtf(fmaxf(var, 0.0f));
    }
}

// ---------------------------------------------------------------------------
// Small dense (f32 in, f32 weights) + silu -> f16 out. Used for encoders.
// ---------------------------------------------------------------------------
__global__ void k_encode(const float* __restrict__ in, const float* __restrict__ w,
                         const float* __restrict__ b, _Float16* __restrict__ out,
                         int ldO, int c0, int R, int K, int Nout) {
    long t = (long)blockIdx.x * blockDim.x + threadIdx.x;
    if (t >= (long)R * Nout) return;
    int r = (int)(t / Nout), n = (int)(t - (long)r * Nout);
    float acc = b[n];
    for (int k = 0; k < K; ++k) acc += in[(long)r * K + k] * w[(long)k * Nout + n];
    out[(long)r * ldO + c0 + n] = (_Float16)silu_f(acc);
}

// ---------------------------------------------------------------------------
// Vectorized gather of 32 f16 columns (v8h granules; optionally double
// indirect: row = idx2[idx1[r]]). All source sub-rows are 16B aligned.
// ---------------------------------------------------------------------------
__global__ void k_gather32(_Float16* __restrict__ dst, int ldD, int c0d,
                           const _Float16* __restrict__ src, int ldS,
                           const int* __restrict__ idx1, const int* __restrict__ idx2,
                           long R) {
    long t = (long)blockIdx.x * blockDim.x + threadIdx.x;
    if (t >= R * 4) return;
    long r = t >> 2;
    int f8 = (int)(t & 3) * 8;
    long sr = idx1 ? (long)idx1[r] : r;
    if (idx2) sr = idx2[sr];
    *(v8h*)(dst + r * ldD + c0d + f8) = *(const v8h*)(src + sr * ldS + f8);
}

// ---------------------------------------------------------------------------
// Segment sum: f16 src [R,F] scattered by idx into f32 sums [Nseg,F]
// ---------------------------------------------------------------------------
__global__ void k_segadd(const _Float16* __restrict__ src, int F,
                         const int* __restrict__ idx, float* __restrict__ sums,
                         long RF) {
    long t = (long)blockIdx.x * blockDim.x + threadIdx.x;
    if (t >= RF) return;
    long r = t / F;
    int f = (int)(t - r * F);
    atomicAdd(&sums[(long)idx[r] * F + f], (float)src[t]);
}

__global__ void k_divcols(const float* __restrict__ sums, const float* __restrict__ cnt,
                          _Float16* __restrict__ dst, int ldD, int c0, int F, int R) {
    long t = (long)blockIdx.x * blockDim.x + threadIdx.x;
    if (t >= (long)R * F) return;
    int r = (int)(t / F), f = (int)(t - (long)r * F);
    float m = sums[t] / fmaxf(cnt[r], 1.0f);
    dst[(long)r * ldD + c0 + f] = (_Float16)m;
}

// ---------------------------------------------------------------------------
// WMMA GEMM: C[M,Nout] = silu(A[M,K] @ W[K,Nout] + bias), f16 in/out,
// f32 accumulate via v_wmma_f32_16x16x32_f16.
//   A  : f16 row-major [M,K] (row stride == K)
//   Wt : f16 TRANSPOSED weights, row-major [Nout,K]
//   C  : f16, row stride ldC
// Each wave computes a 16 x (NT*16) strip: one A-fragment load feeds NT
// WMMAs (A reuse), 8 waves / 256-thread block.
// Requires: M % 16 == 0, K % 32 == 0, Nout in {32,64,128}, NT | (Nout/16).
// ---------------------------------------------------------------------------
template <int NT>
__global__ __launch_bounds__(256) void k_wmma_gemm(
    const _Float16* __restrict__ A, const _Float16* __restrict__ Wt,
    const float* __restrict__ bias, _Float16* __restrict__ C, int ldC,
    int M, int K, int Nout) {
    const int lane = threadIdx.x & 31;
    const int wave = threadIdx.x >> 5;
    const int colTiles = Nout >> 4;
    const int wavesPerStrip = colTiles / NT;              // 1 or 2
    const int stripsPerBlock = 8 / wavesPerStrip;         // 8 or 4
    const int m0 = (blockIdx.x * stripsPerBlock + wave / wavesPerStrip) * 16;
    const int n0 = (wave % wavesPerStrip) * (NT * 16);
    if (m0 >= M) return;                                  // wave-uniform guard

    const int l15 = lane & 15;
    const int lh  = lane >> 4;
    // A fragment (contiguous k order, same as B): lane<16 -> K 0..15,
    // lane>=16 -> K 16..31
    const _Float16* pa = A + (long)(m0 + l15) * K + lh * 16;
    // B fragment (same contiguous k order as A), Wt is [Nout,K]: lane holds column
    // n0+t*16+l15; lane<16 -> K 0..15, lane>=16 -> K 16..31
    const _Float16* pb = Wt + (long)(n0 + l15) * K + lh * 16;

    v8f acc[NT];
    #pragma unroll
    for (int t = 0; t < NT; ++t) { v8f z = {}; acc[t] = z; }

    for (int k0 = 0; k0 < K; k0 += 32) {
        v8h alo = *(const v8h*)(pa + k0);
        v8h ahi = *(const v8h*)(pa + k0 + 8);
        v16h a;
        #pragma unroll
        for (int j = 0; j < 8; ++j) { a[j] = alo[j]; a[8 + j] = ahi[j]; }
        #pragma unroll
        for (int t = 0; t < NT; ++t) {
            v16h b = *(const v16h*)(pb + (long)t * 16 * K + k0);
            acc[t] = __builtin_amdgcn_wmma_f32_16x16x32_f16(
                false, a, false, b, (short)0, acc[t], false, false);
        }
    }

    const int mb = m0 + lh * 8;                           // C layout: lane>=16 -> M+8
    #pragma unroll
    for (int t = 0; t < NT; ++t) {
        const int n = n0 + t * 16 + l15;
        const float bi = bias ? bias[n] : 0.0f;
        #pragma unroll
        for (int r = 0; r < 8; ++r) {
            float v = acc[t][r] + bi;
            v = silu_f(v);
            C[(long)(mb + r) * ldC + n] = (_Float16)v;
        }
    }
}

// ---------------------------------------------------------------------------
// Final tiny dense: out[B,3] = d2[B,32](f16) @ dec_w(f32) + dec_b, no act
// ---------------------------------------------------------------------------
__global__ void k_final(const _Float16* __restrict__ d2, const float* __restrict__ w,
                        const float* __restrict__ b, float* __restrict__ out, int B_) {
    int t = blockIdx.x * blockDim.x + threadIdx.x;
    if (t >= B_ * 3) return;
    int r = t / 3, n = t - r * 3;
    float acc = b[n];
    #pragma unroll
    for (int k = 0; k < 32; ++k) acc += (float)d2[(long)r * 32 + k] * w[k * 3 + n];
    out[t] = acc;
}

// ---------------------------------------------------------------------------
// Host orchestration
// ---------------------------------------------------------------------------
extern "C" void kernel_launch(void* const* d_in, const int* in_sizes, int n_in,
                              void* d_out, int out_size, void* d_ws, size_t ws_size,
                              hipStream_t stream) {
    const int N = in_sizes[3];         // 20000
    const int E = in_sizes[2] / 2;     // 200000
    const int Bc = 128, H = 32, Lc = 10;

    // ---- inputs ----
    const float* x        = (const float*)d_in[0];
    const float* eattr    = (const float*)d_in[1];
    const int*   eidx     = (const int*)d_in[2];
    const int*   row      = eidx;
    const int*   col      = eidx + E;
    const int*   batch    = (const int*)d_in[3];
    const float* xew = (const float*)d_in[4];  const float* xeb = (const float*)d_in[5];
    const float* eew = (const float*)d_in[6];  const float* eeb = (const float*)d_in[7];
    const float* uew = (const float*)d_in[8];  const float* ueb = (const float*)d_in[9];
    const float* em_w  = (const float*)d_in[10]; const float* em_b  = (const float*)d_in[11];
    const float* em_dw = (const float*)d_in[12]; const float* em_db = (const float*)d_in[13];
    const float* nm1_w = (const float*)d_in[14]; const float* nm1_b = (const float*)d_in[15];
    const float* nm2_w = (const float*)d_in[16]; const float* nm2_b = (const float*)d_in[17];
    const float* nm_dw = (const float*)d_in[18]; const float* nm_db = (const float*)d_in[19];
    const float* gm_w  = (const float*)d_in[20]; const float* gm_b  = (const float*)d_in[21];
    const float* gm_dw = (const float*)d_in[22]; const float* gm_db = (const float*)d_in[23];
    const float* d1w = (const float*)d_in[24]; const float* d1b = (const float*)d_in[25];
    const float* d2w = (const float*)d_in[26]; const float* d2b = (const float*)d_in[27];
    const float* dw  = (const float*)d_in[28]; const float* db  = (const float*)d_in[29];

    // ---- workspace carve-up ----
    char* wp = (char*)d_ws;
    auto alloc = [&](size_t bytes) -> char* {
        char* p = wp;
        wp += (bytes + 255) & ~(size_t)255;
        return p;
    };
    _Float16* wt_em  = (_Float16*)alloc((size_t)Lc * 2 * 128 * 128 * 2);
    _Float16* wt_emd = (_Float16*)alloc((size_t)Lc * 128 * 32 * 2);
    _Float16* wt_nm1 = (_Float16*)alloc((size_t)Lc * 2 * 64 * 64 * 2);
    _Float16* wt_nm2 = (_Float16*)alloc((size_t)Lc * 2 * 128 * 128 * 2);
    _Float16* wt_nmd = (_Float16*)alloc((size_t)Lc * 128 * 32 * 2);
    _Float16* wt_gm  = (_Float16*)alloc((size_t)Lc * 2 * 64 * 64 * 2);
    _Float16* wt_gmd = (_Float16*)alloc((size_t)Lc * 64 * 32 * 2);
    _Float16* wt_d1  = (_Float16*)alloc((size_t)320 * 64 * 2);
    _Float16* wt_d2  = (_Float16*)alloc((size_t)128 * 32 * 2);
    float* esum  = (float*)alloc((size_t)N * 6 * 4);
    float* esq   = (float*)alloc((size_t)N * 6 * 4);
    float* deg   = (float*)alloc((size_t)N * 4);
    float* x0    = (float*)alloc((size_t)N * 17 * 4);
    float* usum  = (float*)alloc((size_t)Bc * 17 * 4);
    float* usq   = (float*)alloc((size_t)Bc * 17 * 4);
    float* bcnt  = (float*)alloc((size_t)Bc * 4);
    float* u0    = (float*)alloc((size_t)Bc * 34 * 4);
    _Float16* hx16 = (_Float16*)alloc((size_t)N * 32 * 2);
    _Float16* he16 = (_Float16*)alloc((size_t)E * 32 * 2);
    _Float16* hu0  = (_Float16*)alloc((size_t)Bc * 32 * 2);
    _Float16* us   = (_Float16*)alloc((size_t)Bc * 320 * 2);
    _Float16* big1 = (_Float16*)alloc((size_t)E * 128 * 2);
    _Float16* big2 = (_Float16*)alloc((size_t)E * 128 * 2);
    _Float16* nod1 = (_Float16*)alloc((size_t)N * 128 * 2);
    _Float16* nod2 = (_Float16*)alloc((size_t)N * 128 * 2);
    float* nsums   = (float*)alloc((size_t)N * 64 * 4);
    float* bsums   = (float*)alloc((size_t)Bc * 32 * 4);
    float* bsums64 = (float*)alloc((size_t)Bc * 64 * 4);
    _Float16* gbuf = (_Float16*)alloc((size_t)Bc * 64 * 2);
    _Float16* gtmp = (_Float16*)alloc((size_t)Bc * 64 * 2);
    _Float16* dbuf = (_Float16*)alloc((size_t)Bc * 128 * 2);
    _Float16* d2o  = (_Float16*)alloc((size_t)Bc * 32 * 2);
    (void)ws_size; (void)n_in; (void)out_size;

    auto cdiv = [](long a, long b) -> int { return (int)((a + b - 1) / b); };
    const int T = 256;

    // ---- weight convert + transpose (f32 [K,N] -> f16 [N,K]) ----
    auto castT = [&](const float* W, _Float16* Wt, int K, int Nn, int nm) {
        long tot = (long)nm * K * Nn;
        k_castT<<<cdiv(tot, T), T, 0, stream>>>(W, Wt, K, Nn, nm);
    };
    castT(em_w,  wt_em,  128, 128, Lc * 2);
    castT(em_dw, wt_emd, 128, 32,  Lc);
    castT(nm1_w, wt_nm1, 64,  64,  Lc * 2);
    castT(nm2_w, wt_nm2, 128, 128, Lc * 2);
    castT(nm_dw, wt_nmd, 128, 32,  Lc);
    castT(gm_w,  wt_gm,  64,  64,  Lc * 2);
    castT(gm_dw, wt_gmd, 64,  32,  Lc);
    castT(d1w,   wt_d1,  320, 64,  1);
    castT(d2w,   wt_d2,  128, 32,  1);

    auto gemm = [&](const _Float16* A, const _Float16* Wt, const float* bias,
                    _Float16* C, int ldC, int M, int K, int Nout) {
        const int colTiles = Nout >> 4;
        const int NT = (Nout >= 64) ? 4 : 2;
        const int rowsPerBlock = (8 / (colTiles / NT)) * 16;  // 64 or 128
        const int grid = cdiv(M, rowsPerBlock);
        if (NT == 4)
            k_wmma_gemm<4><<<grid, T, 0, stream>>>(A, Wt, bias, C, ldC, M, K, Nout);
        else
            k_wmma_gemm<2><<<grid, T, 0, stream>>>(A, Wt, bias, C, ldC, M, K, Nout);
    };
    auto gather = [&](_Float16* dst, int ldD, int c0d, const _Float16* src, int ldS,
                      const int* i1, const int* i2, long R) {
        k_gather32<<<cdiv(R * 4, T), T, 0, stream>>>(dst, ldD, c0d, src, ldS, i1, i2, R);
    };

    // ---- feature augmentation ----
    hipMemsetAsync(esum, 0, (size_t)N * 6 * 4, stream);
    hipMemsetAsync(esq,  0, (size_t)N * 6 * 4, stream);
    hipMemsetAsync(deg,  0, (size_t)N * 4, stream);
    hipMemsetAsync(usum, 0, (size_t)Bc * 17 * 4, stream);
    hipMemsetAsync(usq,  0, (size_t)Bc * 17 * 4, stream);
    hipMemsetAsync(bcnt, 0, (size_t)Bc * 4, stream);
    k_estats<<<cdiv(E, T), T, 0, stream>>>(eattr, col, esum, esq, deg, E);
    k_x0final<<<cdiv(N, T), T, 0, stream>>>(x, esum, esq, deg, x0, N);
    k_ustats<<<cdiv(N, T), T, 0, stream>>>(x0, batch, usum, usq, bcnt, N);
    k_u0final<<<cdiv(Bc, T), T, 0, stream>>>(usum, usq, bcnt, u0, Bc);

    // ---- encoders ----
    k_encode<<<cdiv((long)N * 32, T), T, 0, stream>>>(x0, xew, xeb, hx16, 32, 0, N, 17, 32);
    k_encode<<<cdiv((long)E * 32, T), T, 0, stream>>>(eattr, eew, eeb, he16, 32, 0, E, 6, 32);
    k_encode<<<cdiv((long)Bc * 32, T), T, 0, stream>>>(u0, uew, ueb, hu0, 32, 0, Bc, 34, 32);

    // ---- message-passing layers ----
    for (int i = 0; i < Lc; ++i) {
        const _Float16* hu = (i == 0) ? hu0 : (us + (i - 1) * H);
        const int ldHU = (i == 0) ? H : (Lc * H);

        // EdgeModel: A_e = [h_x[row] | h_x[col] | h_e | h_u[batch[row]]]  [E,128]
        gather(big1, 128, 0,  hx16, 32, row, nullptr, E);
        gather(big1, 128, 32, hx16, 32, col, nullptr, E);
        gather(big1, 128, 64, he16, 32, nullptr, nullptr, E);
        gather(big1, 128, 96, hu, ldHU, row, batch, E);
        gemm(big1, wt_em + (size_t)(i * 2 + 0) * 16384, em_b + (i * 2 + 0) * 128, big2, 128, E, 128, 128);
        gemm(big2, wt_em + (size_t)(i * 2 + 1) * 16384, em_b + (i * 2 + 1) * 128, big1, 128, E, 128, 128);
        gemm(big1, wt_emd + (size_t)i * 4096, em_db + i * 32, he16, 32, E, 128, 32);

        // NodeModel edge MLP: A = [h_x[row] | h_e]  [E,64]
        gather(big1, 64, 0,  hx16, 32, row, nullptr, E);
        gather(big1, 64, 32, he16, 32, nullptr, nullptr, E);
        gemm(big1, wt_nm1 + (size_t)(i * 2 + 0) * 4096, nm1_b + (i * 2 + 0) * 64, big2, 64, E, 64, 64);
        gemm(big2, wt_nm1 + (size_t)(i * 2 + 1) * 4096, nm1_b + (i * 2 + 1) * 64, big1, 64, E, 64, 64);
        hipMemsetAsync(nsums, 0, (size_t)N * 64 * 4, stream);
        k_segadd<<<cdiv((long)E * 64, T), T, 0, stream>>>(big1, 64, col, nsums, (long)E * 64);

        // NodeModel node MLP: A = [h_x | agg | h_u[batch]]  [N,128]
        gather(nod1, 128, 0, hx16, 32, nullptr, nullptr, N);
        k_divcols<<<cdiv((long)N * 64, T), T, 0, stream>>>(nsums, deg, nod1, 128, 32, 64, N);
        gather(nod1, 128, 96, hu, ldHU, batch, nullptr, N);
        gemm(nod1, wt_nm2 + (size_t)(i * 2 + 0) * 16384, nm2_b + (i * 2 + 0) * 128, nod2, 128, N, 128, 128);
        gemm(nod2, wt_nm2 + (size_t)(i * 2 + 1) * 16384, nm2_b + (i * 2 + 1) * 128, nod1, 128, N, 128, 128);
        gemm(nod1, wt_nmd + (size_t)i * 4096, nm_db + i * 32, hx16, 32, N, 128, 32);

        // GlobalModel: A = [h_u | seg_mean(h_x, batch)]  [B,64]
        hipMemsetAsync(bsums, 0, (size_t)Bc * 32 * 4, stream);
        k_segadd<<<cdiv((long)N * 32, T), T, 0, stream>>>(hx16, 32, batch, bsums, (long)N * 32);
        gather(gbuf, 64, 0, hu, ldHU, nullptr, nullptr, Bc);
        k_divcols<<<cdiv((long)Bc * 32, T), T, 0, stream>>>(bsums, bcnt, gbuf, 64, 32, 32, Bc);
        gemm(gbuf, wt_gm + (size_t)(i * 2 + 0) * 4096, gm_b + (i * 2 + 0) * 64, gtmp, 64, Bc, 64, 64);
        gemm(gtmp, wt_gm + (size_t)(i * 2 + 1) * 4096, gm_b + (i * 2 + 1) * 64, gbuf, 64, Bc, 64, 64);
        // write new h_u directly into the [B, 10H] concat buffer `us`
        gemm(gbuf, wt_gmd + (size_t)i * 2048, gm_db + i * 32, us + i * H, Lc * H, Bc, 64, 32);
    }

    // ---- decoder ----
    // dec1: silu(us @ dec1_w + b) -> dbuf[:, 0:64]
    gemm(us, wt_d1, d1b, dbuf, 128, Bc, 320, 64);
    // node_out = [h_x | seg_mean(h_e, col)]  [N,64]
    gather(nod1, 64, 0, hx16, 32, nullptr, nullptr, N);
    hipMemsetAsync(nsums, 0, (size_t)N * 32 * 4, stream);
    k_segadd<<<cdiv((long)E * 32, T), T, 0, stream>>>(he16, 32, col, nsums, (long)E * 32);
    k_divcols<<<cdiv((long)N * 32, T), T, 0, stream>>>(nsums, deg, nod1, 64, 32, 32, N);
    // seg_mean(node_out, batch) -> dbuf[:, 64:128]
    hipMemsetAsync(bsums64, 0, (size_t)Bc * 64 * 4, stream);
    k_segadd<<<cdiv((long)N * 64, T), T, 0, stream>>>(nod1, 64, batch, bsums64, (long)N * 64);
    k_divcols<<<cdiv((long)Bc * 64, T), T, 0, stream>>>(bsums64, bcnt, dbuf, 128, 64, 64, Bc);
    // dec2 + final projection
    gemm(dbuf, wt_d2, d2b, d2o, 32, Bc, 128, 32);
    k_final<<<cdiv(Bc * 3, T), T, 0, stream>>>(d2o, dw, db, (float*)d_out, Bc);
}
